// SelfAttention_47115791237225
// MI455X (gfx1250) — matrix operand order repaired and checked by reading, not compiled, not run on hardware
//
#include <hip/hip_runtime.h>

#define DIMS  1024
#define SEQ   2048
#define BATCH 4

typedef __attribute__((ext_vector_type(16))) _Float16 v16h;
typedef __attribute__((ext_vector_type(8)))  _Float16 v8h;
typedef __attribute__((ext_vector_type(8)))  float    v8f;
typedef __attribute__((ext_vector_type(4)))  int      v4i;

union FragH { v16h v; v8h h[2]; };

// ---------------- CDNA5 async global->LDS copy (ASYNCcnt path) ----------------
#if __has_builtin(__builtin_amdgcn_global_load_async_to_lds_b128)
#define HAS_ASYNC_LDS 1
#else
#define HAS_ASYNC_LDS 0
#endif

typedef __attribute__((address_space(1))) v4i g_v4i;   // global int4
typedef __attribute__((address_space(3))) v4i l_v4i;   // LDS int4

__device__ __forceinline__ void cp16_g2l(const _Float16* g, _Float16* l) {
#if HAS_ASYNC_LDS
  __builtin_amdgcn_global_load_async_to_lds_b128(
      (g_v4i*)g, (l_v4i*)l, /*offset=*/0, /*cpol=*/0);
#else
  *(v8h*)l = *(const v8h*)g;   // global_load_b128 + ds_store_b128 fallback
#endif
}

template <int N>
__device__ __forceinline__ void wait_async() {
#if HAS_ASYNC_LDS
#if __has_builtin(__builtin_amdgcn_s_wait_asynccnt)
  __builtin_amdgcn_s_wait_asynccnt(N);
#else
  asm volatile("s_wait_asynccnt %0" ::"i"(N) : "memory");
#endif
#endif
}

// ---------------------------------------------------------------- convert
__global__ __launch_bounds__(256)
void cvt_f32_f16(const float* __restrict__ in, _Float16* __restrict__ out, int n) {
  int i = blockIdx.x * blockDim.x + threadIdx.x;
  int stride = gridDim.x * blockDim.x;
  for (; i < n; i += stride) out[i] = (_Float16)in[i];
}

// ---------------------------------------------------------------- GEMM
// C[m][n] = sum_k A[m][k] * B[n][k]   (both operands row-major, K contiguous)
// OUT_MODE: 0 = f32 row-major, 1 = f16 row-major, 2 = f16 transposed (C[n][m])
constexpr int BM = 128, BN = 128, BK = 32, LDSH = 40;
constexpr int ABUF = BM * LDSH;          // 5120 halves per matrix per buffer
constexpr int BUFSZ = 2 * ABUF;          // A tile + B tile

template <int OUT_MODE, bool CAUSAL>
__global__ __launch_bounds__(256)
void gemm_f16(const _Float16* __restrict__ A, const _Float16* __restrict__ B,
              void* __restrict__ Cv, int K, int lda, int ldb, int ldc,
              long strideA, long strideB, long strideC)
{
  const int m0 = blockIdx.y * BM;
  const int n0 = blockIdx.x * BN;
  if (CAUSAL && n0 > m0 + BM - 1) return;   // tile fully above causal diagonal

  A += (long)blockIdx.z * strideA;
  B += (long)blockIdx.z * strideB;

  // double-buffered LDS: [buf][A tile | B tile]
  __shared__ __align__(16) _Float16 smem[2 * BUFSZ];

  const int tid  = threadIdx.x;
  const int lane = tid & 31;
  const int lr   = lane & 15;
  const int lhi  = lane >> 4;
  const int wave = tid >> 5;
  const int wm   = (wave >> 1) * 32;   // wave M offset in block tile
  const int wn   = (wave & 1) * 64;    // wave N offset in block tile

  // ---- static per-thread staging assignment: 2 chunks of A + 2 of B per tile
  const int srow = tid >> 2;           // 0..63
  const int scol = (tid & 3) * 8;      // 0,8,16,24
  const long gA2 = (long)64 * lda;     // second chunk: row + 64
  const long gB2 = (long)64 * ldb;
  const int  lofs = srow * LDSH + scol;

  const _Float16* pA = A + (long)(m0 + srow) * lda + scol;   // running prefetch ptrs
  const _Float16* pB = B + (long)(n0 + srow) * ldb + scol;

  auto issue = [&](int bufi, const _Float16* qA, const _Float16* qB) {
    _Float16* la = &smem[bufi * BUFSZ] + lofs;
    _Float16* lb = la + ABUF;
    cp16_g2l(qA,       la);
    cp16_g2l(qA + gA2, la + 64 * LDSH);
    cp16_g2l(qB,       lb);
    cp16_g2l(qB + gB2, lb + 64 * LDSH);
  };

  v8f acc[2][4] = {};                  // 32x64 f32 accumulator per wave

  auto compute = [&](const _Float16* Ash) {
    const _Float16* Bsh = Ash + ABUF;
    // ---- A fragments: interleaved K-halves per ISA 16-bit A layout
    FragH a[2], b[4];
#pragma unroll
    for (int i = 0; i < 2; ++i) {
      const _Float16* p = &Ash[(wm + i * 16 + lr) * LDSH + lhi * 8];
      a[i].h[0] = *(const v8h*)p;          // K = lhi*8 .. +7
      a[i].h[1] = *(const v8h*)(p + 16);   // K = 16+lhi*8 .. +7
    }
    // ---- B fragments: interleaved K-halves matching A layout
#pragma unroll
    for (int j = 0; j < 4; ++j) {
      const _Float16* p = &Bsh[(wn + j * 16 + lr) * LDSH + lhi * 8];
      b[j].h[0] = *(const v8h*)p;          // K = lhi*8 .. +7
      b[j].h[1] = *(const v8h*)(p + 16);   // K = 16+lhi*8 .. +7
    }
#pragma unroll
    for (int i = 0; i < 2; ++i)
#pragma unroll
      for (int j = 0; j < 4; ++j)
        acc[i][j] = __builtin_amdgcn_wmma_f32_16x16x32_f16(
            false, a[i].v, false, b[j].v, (short)0, acc[i][j], false, false);
  };

  issue(0, pA, pB);                    // prologue: stage first tile
  int cur = 0;

  // steady state: branch-free body, unconditional prefetch of next tile
  for (int kt = 0; kt < K - BK; kt += BK) {
    pA += BK;
    pB += BK;
    issue(cur ^ 1, pA, pB);            // prefetch next tile into other buffer
    wait_async<4>();                   // 4 newest in flight -> current tile done
    __syncthreads();                   // all waves' current-tile data visible
    compute(&smem[cur * BUFSZ]);
    __syncthreads();                   // protect buffer before it is re-filled
    cur ^= 1;
  }

  // peeled last tile: nothing left to prefetch
  wait_async<0>();
  __syncthreads();
  compute(&smem[cur * BUFSZ]);

  // ---- epilogue: C/D layout = VGPR v -> row v + 8*lanehi, col = lane&15
#pragma unroll
  for (int i = 0; i < 2; ++i)
#pragma unroll
    for (int j = 0; j < 4; ++j)
#pragma unroll
      for (int v = 0; v < 8; ++v) {
        const int row = m0 + wm + i * 16 + v + lhi * 8;
        const int col = n0 + wn + j * 16 + lr;
        if (OUT_MODE == 0)
          ((float*)Cv)[(long)blockIdx.z * strideC + (long)row * ldc + col] = acc[i][j][v];
        else if (OUT_MODE == 1)
          ((_Float16*)Cv)[(long)blockIdx.z * strideC + (long)row * ldc + col] =
              (_Float16)acc[i][j][v];
        else
          ((_Float16*)Cv)[(long)blockIdx.z * strideC + (long)col * ldc + row] =
              (_Float16)acc[i][j][v];
      }
}

// ---------------------------------------------------------------- softmax
__global__ __launch_bounds__(256)
void softmax_causal(const float* __restrict__ S, _Float16* __restrict__ P) {
  const int row = blockIdx.x;           // 0 .. BATCH*SEQ-1
  const int q   = row & (SEQ - 1);
  const float* s = S + (long)row * SEQ;
  _Float16*    p = P + (long)row * SEQ;
  const int n   = q + 1;                // valid (unmasked) length
  const int tid = threadIdx.x;
  __shared__ float red[256];

  float m = -3.402823466e38f;
  for (int k = tid; k < n; k += 256) m = fmaxf(m, s[k]);
  red[tid] = m; __syncthreads();
  for (int off = 128; off > 0; off >>= 1) {
    if (tid < off) red[tid] = fmaxf(red[tid], red[tid + off]);
    __syncthreads();
  }
  m = red[0]; __syncthreads();

  float sum = 0.f;
  for (int k = tid; k < n; k += 256) sum += __expf(s[k] - m);
  red[tid] = sum; __syncthreads();
  for (int off = 128; off > 0; off >>= 1) {
    if (tid < off) red[tid] += red[tid + off];
    __syncthreads();
  }
  const float inv = 1.0f / red[0];

  for (int k = tid; k < SEQ; k += 256) {
    float v = (k < n) ? __expf(s[k] - m) * inv : 0.0f;
    p[k] = (_Float16)v;
  }
}

// ---------------------------------------------------------------- launcher
extern "C" void kernel_launch(void* const* d_in, const int* in_sizes, int n_in,
                              void* d_out, int out_size, void* d_ws, size_t ws_size,
                              hipStream_t stream) {
  (void)in_sizes; (void)n_in; (void)out_size; (void)ws_size;
  const float* x  = (const float*)d_in[0];
  const float* Wq = (const float*)d_in[1];
  const float* Wk = (const float*)d_in[2];
  const float* Wv = (const float*)d_in[3];
  float* out = (float*)d_out;
  char*  ws  = (char*)d_ws;

  const size_t MB = (size_t)1 << 20;
  _Float16* Wqh = (_Float16*)(ws + 0 * MB);   //  2 MB
  _Float16* Wkh = (_Float16*)(ws + 2 * MB);   //  2 MB
  _Float16* Wvh = (_Float16*)(ws + 4 * MB);   //  2 MB
  _Float16* Xh  = (_Float16*)(ws + 6 * MB);   // 16 MB
  _Float16* Qh  = (_Float16*)(ws + 22 * MB);  // 16 MB
  _Float16* Kh  = (_Float16*)(ws + 38 * MB);  // 16 MB
  _Float16* Vt  = (_Float16*)(ws + 54 * MB);  // 16 MB  ([e][b*SEQ+s])
  float*    Sc  = (float*)   (ws + 70 * MB);  // 64 MB
  _Float16* P   = (_Float16*)(ws + 0 * MB);   // 32 MB, aliases Wh/Xh/part-of-Qh
                                              // (their consumers finish first)

  const int nX = BATCH * SEQ * DIMS;
  const int nW = DIMS * DIMS;
  const int MTOK = BATCH * SEQ;               // 8192 tokens

  cvt_f32_f16<<<1024, 256, 0, stream>>>(x,  Xh,  nX);
  cvt_f32_f16<<<512,  256, 0, stream>>>(Wq, Wqh, nW);
  cvt_f32_f16<<<512,  256, 0, stream>>>(Wk, Wkh, nW);
  cvt_f32_f16<<<512,  256, 0, stream>>>(Wv, Wvh, nW);

  dim3 blk(256);

  // QKV projections: [8192 x 1024] = Xh[8192 x 1024] * W[1024 x 1024]^T
  dim3 gp(DIMS / BN, MTOK / BM, 1);
  gemm_f16<1, false><<<gp, blk, 0, stream>>>(Xh, Wqh, Qh, DIMS, DIMS, DIMS, DIMS, 0, 0, 0);
  gemm_f16<1, false><<<gp, blk, 0, stream>>>(Xh, Wkh, Kh, DIMS, DIMS, DIMS, DIMS, 0, 0, 0);
  gemm_f16<2, false><<<gp, blk, 0, stream>>>(Xh, Wvh, Vt, DIMS, DIMS, DIMS, MTOK, 0, 0, 0);

  // scores per batch: S[q][k] = Q[q]·K[k]   (skip tiles above diagonal)
  dim3 gs(SEQ / BN, SEQ / BM, BATCH);
  gemm_f16<0, true><<<gs, blk, 0, stream>>>(Qh, Kh, Sc, DIMS, DIMS, DIMS, SEQ,
                                            (long)SEQ * DIMS, (long)SEQ * DIMS,
                                            (long)SEQ * SEQ);

  // causal softmax rows -> P (f16), masked tail zeroed
  softmax_causal<<<BATCH * SEQ, 256, 0, stream>>>(Sc, P);

  // output per batch: O[q][e] = sum_k P[q][k] * Vt[e][b*SEQ+k]
  dim3 go(DIMS / BN, SEQ / BM, BATCH);
  gemm_f16<0, false><<<go, blk, 0, stream>>>(P, Vt, out, SEQ, SEQ, MTOK, DIMS,
                                             (long)SEQ * SEQ, (long)SEQ,
                                             (long)SEQ * DIMS);
}
